// CortexReasoner_67860483276968
// MI455X (gfx1250) — matrix operand order repaired and checked by reading, not compiled, not run on hardware
//
#include <hip/hip_runtime.h>
#include <hip/hip_bf16.h>
#include <math.h>

typedef __attribute__((ext_vector_type(16))) _Float16 v16h;
typedef __attribute__((ext_vector_type(8)))  _Float16 v8h;
typedef __attribute__((ext_vector_type(8)))  float    v8f;

#define B_   2
#define T_   2048
#define D_   1024
#define NH   8
#define HD   128
#define RM   64
#define FBM  32
#define BT   (B_*T_)
#define QEXT 192        // 128 rope'd dims + 64 fourier features
#define KSTR 200        // ldsK row stride (halfs): 400B, conflict-free, 16B aligned
#define VSTR 40         // ldsV row stride (halfs): 80B, conflict-free, 16B aligned

union FragH { v16h v; v8h h[2]; };
union FragF { v8f  v; float f[8]; };

// ---- WMMA fragment loaders (16x16x32 f16 layouts per CDNA5 ISA 7.12.2) ----
// A-matrix 16x32: lanes 0-15 -> row=lane, K {0..7,16..23}; lanes 16-31 -> K {8..15,24..31}
__device__ inline v16h load_frag_a(const _Float16* base, int ld) {
  const int lane = threadIdx.x & 31;
  const int hsel = lane >> 4;
  const _Float16* p = base + (size_t)(lane & 15) * ld + hsel * 8;
  FragH f;
  f.h[0] = *(const v8h*)(p);
  f.h[1] = *(const v8h*)(p + 16);
  return f.v;
}
// B-matrix 32x16 loaded column-major == rows of B^T. lanes 0-15: col=lane, K {0..7,16..23};
// lanes 16-31: K {8..15,24..31}.  `base` points at B^T (N x K row-major) tile origin.
__device__ inline v16h load_frag_b(const _Float16* base, int ld) {
  const int lane = threadIdx.x & 31;
  const int hsel = lane >> 4;
  const _Float16* p = base + (size_t)(lane & 15) * ld + hsel * 8;
  FragH f;
  f.h[0] = *(const v8h*)(p);
  f.h[1] = *(const v8h*)(p + 16);
  return f.v;
}

// ---- CDNA5 async copy: 16B per lane, global -> LDS, tracked by ASYNCcnt ----
__device__ inline void async_copy_b128(const _Float16* gsrc, _Float16* ldst) {
  unsigned lds_addr = (unsigned)(uintptr_t)ldst;   // LDS aperture: addr[31:0] = LDS offset
  asm volatile("global_load_async_to_lds_b128 %0, %1, off"
               :: "v"(lds_addr), "v"(gsrc) : "memory");
}
__device__ inline void wait_asynccnt0() {
  asm volatile("s_wait_asynccnt 0x0" ::: "memory");
}

// ---- weight convert + transpose: src (K x N) f32 -> dst (N x K) f16 ----
__global__ void transpose_f16_kernel(const float* __restrict__ src,
                                     _Float16* __restrict__ dst, int K, int N) {
  int idx = blockIdx.x * blockDim.x + threadIdx.x;
  if (idx >= K * N) return;
  int n = idx / K, k = idx % K;
  dst[idx] = (_Float16)src[(size_t)k * N + n];
}

// ---- generic tiled WMMA GEMM: C(MxN) = A(MxK,f32) * Bt(NxK,f16)^T + bias ----
// EPI==1: masked residual epilogue (output projection)
template <int EPI>
__global__ __launch_bounds__(128) void gemm_wmma_kernel(
    const float* __restrict__ A, const _Float16* __restrict__ Bt,
    const float* __restrict__ bias, float* __restrict__ C,
    int M, int N, int K,
    const unsigned char* __restrict__ mask, const float* __restrict__ xres) {
  __shared__ alignas(16) _Float16 ldsA[64 * 48];
  __shared__ alignas(16) _Float16 ldsB[64 * 48];
  const int tid = threadIdx.x;
  const int wid = tid >> 5, lane = tid & 31;
  const int m0 = blockIdx.y * 64, n0 = blockIdx.x * 64;
  const int wm = (wid >> 1) * 32, wn = (wid & 1) * 32;

  v8f c00 = {}, c01 = {}, c10 = {}, c11 = {};

  const int sr = tid >> 1, sc = (tid & 1) * 16;   // staging: 64 rows x 32 cols / 128 thr
  for (int k0 = 0; k0 < K; k0 += 32) {
    { // stage A tile, f32 -> f16  (+ prefetch next tile -> global_prefetch_b8)
      const float* s = A + (size_t)(m0 + sr) * K + k0 + sc;
      if (k0 + 32 < K) __builtin_prefetch(s + 32, 0, 0);
      _Float16* d = &ldsA[sr * 48 + sc];
#pragma unroll
      for (int i = 0; i < 16; ++i) d[i] = (_Float16)s[i];
    }
    { // stage B^T tile (already f16)
      const _Float16* s = Bt + (size_t)(n0 + sr) * K + k0 + sc;
      if (k0 + 32 < K) __builtin_prefetch(s + 32, 0, 0);
      _Float16* d = &ldsB[sr * 48 + sc];
      *(v8h*)(d)     = *(const v8h*)(s);
      *(v8h*)(d + 8) = *(const v8h*)(s + 8);
    }
    __syncthreads();
    v16h a0 = load_frag_a(&ldsA[(wm)      * 48], 48);
    v16h a1 = load_frag_a(&ldsA[(wm + 16) * 48], 48);
    v16h b0 = load_frag_b(&ldsB[(wn)      * 48], 48);
    v16h b1 = load_frag_b(&ldsB[(wn + 16) * 48], 48);
    c00 = __builtin_amdgcn_wmma_f32_16x16x32_f16(false, a0, false, b0, (short)0, c00, false, false);
    c01 = __builtin_amdgcn_wmma_f32_16x16x32_f16(false, a0, false, b1, (short)0, c01, false, false);
    c10 = __builtin_amdgcn_wmma_f32_16x16x32_f16(false, a1, false, b0, (short)0, c10, false, false);
    c11 = __builtin_amdgcn_wmma_f32_16x16x32_f16(false, a1, false, b1, (short)0, c11, false, false);
    __syncthreads();
  }

  const int colL = lane & 15, halfL = lane >> 4;
  v8f frags[4] = {c00, c01, c10, c11};
#pragma unroll
  for (int q = 0; q < 4; ++q) {
    FragF cf; cf.v = frags[q];
    int rbase = m0 + wm + (q >> 1) * 16 + 8 * halfL;
    int col   = n0 + wn + (q & 1) * 16 + colL;
    float bv = bias[col];
#pragma unroll
    for (int r = 0; r < 8; ++r) {
      int row = rbase + r;
      float val = cf.f[r] + bv;
      if (EPI == 1) val = mask[row] ? val : xres[(size_t)row * N + col];
      C[(size_t)row * N + col] = val;
    }
  }
}

// ---- RoPE + Fourier-feature extension + V transpose ----
// qkv (B,T,3D) f32 -> q_ext/k_ext (B,H,T,192) f16, v_t (B,H,HD,T) f16
__global__ __launch_bounds__(64) void rope_extend_kernel(
    const float* __restrict__ qkv, const float* __restrict__ coords,
    const float* __restrict__ W_rope, const float* __restrict__ W_fb,
    const float* __restrict__ beta_cos, const float* __restrict__ beta_sin,
    _Float16* __restrict__ q_ext, _Float16* __restrict__ k_ext,
    _Float16* __restrict__ v_t) {
  const int bt = blockIdx.x;
  const int b = bt / T_, t = bt % T_;
  const int m = threadIdx.x;              // 0..63 : rope pair index
  const float pos = coords[bt];
  const float rs = 0.08838834764831845f;  // 1/sqrt(128)

  float sth, cth;
  __sincosf(pos * W_rope[m], &sth, &cth);

  const float* qrow = qkv + (size_t)bt * (3 * D_);
  const float* krow = qrow + D_;
  const float* vrow = qrow + 2 * D_;

#pragma unroll
  for (int h = 0; h < NH; ++h) {
    const float* qh = qrow + h * HD;
    const float* kh = krow + h * HD;
    float q0 = qh[2 * m], q1 = qh[2 * m + 1];
    float k0 = kh[2 * m], k1 = kh[2 * m + 1];
    size_t eb = ((size_t)(b * NH + h) * T_ + t) * QEXT;
    q_ext[eb + 2 * m]     = (_Float16)((q0 * cth - q1 * sth) * rs);
    q_ext[eb + 2 * m + 1] = (_Float16)((q0 * sth + q1 * cth) * rs);
    k_ext[eb + 2 * m]     = (_Float16)(k0 * cth - k1 * sth);
    k_ext[eb + 2 * m + 1] = (_Float16)(k0 * sth + k1 * cth);
    // V transpose: [b,h,d,t]
    size_t vb = (size_t)(b * NH + h) * HD * T_;
    v_t[vb + (size_t)m * T_ + t]        = (_Float16)vrow[h * HD + m];
    v_t[vb + (size_t)(m + 64) * T_ + t] = (_Float16)vrow[h * HD + m + 64];
  }
  if (m < FBM) {
    float sS, cS;
    __sincosf(pos * W_fb[m], &sS, &cS);
    float bc = beta_cos[m], bs = beta_sin[m];
    float fq0 = cS * bc + sS * bs;
    float fq1 = sS * bc - cS * bs;
#pragma unroll
    for (int h = 0; h < NH; ++h) {
      size_t eb = ((size_t)(b * NH + h) * T_ + t) * QEXT + HD;
      q_ext[eb + m]       = (_Float16)fq0;
      q_ext[eb + FBM + m] = (_Float16)fq1;
      k_ext[eb + m]       = (_Float16)cS;
      k_ext[eb + FBM + m] = (_Float16)sS;
    }
  }
}

// ---- flash attention: 8 waves/block share (b,h); K/V tiles async-staged to LDS ----
// Each wave owns 16 q-rows; 32-key tiles; online softmax; bias folded via K=192.
__global__ __launch_bounds__(256) void flash_attn_kernel(
    const _Float16* __restrict__ q_ext, const _Float16* __restrict__ k_ext,
    const _Float16* __restrict__ v_t, float* __restrict__ y) {
  __shared__ alignas(16) _Float16 ldsK[32 * KSTR];   // 32 keys x 192 feats
  __shared__ alignas(16) _Float16 ldsV[HD * VSTR];   // 128 dims x 32 keys
  __shared__ alignas(16) _Float16 ldsP[8][16 * 32];  // per-wave P re-layout tile
  const int tid = threadIdx.x;
  const int wid = tid >> 5, lane = tid & 31;
  const int bh = blockIdx.x;
  const int b = bh >> 3, h = bh & 7;
  const int q0 = (blockIdx.y * 8 + wid) * 16;
  const int colL = lane & 15, halfL = lane >> 4;

  const _Float16* qb = q_ext + ((size_t)bh * T_ + q0) * QEXT;
  const _Float16* kb = k_ext + (size_t)bh * T_ * QEXT;
  const _Float16* vb = v_t   + (size_t)bh * HD * T_;

  v16h qf[6];
#pragma unroll
  for (int j = 0; j < 6; ++j) qf[j] = load_frag_a(qb + j * 32, QEXT);

  v8f zf = {};
  v8f acc[8];
#pragma unroll
  for (int n = 0; n < 8; ++n) acc[n] = zf;
  float mrun[8], lrun[8];
#pragma unroll
  for (int r = 0; r < 8; ++r) { mrun[r] = -1e30f; lrun[r] = 0.f; }

  _Float16* Pl = ldsP[wid];

  for (int kt = 0; kt < T_; kt += 32) {
    // ---- cooperative async staging: K tile 32x192 (768 16B chunks / 256 thr = 3),
    //      V tile 128x32 (512 chunks / 256 thr = 2) ----
#pragma unroll
    for (int i = 0; i < 3; ++i) {
      int c = tid + 256 * i;             // chunk id
      int row = c / 24, c8 = c % 24;     // 24 chunks of 8 halfs per 192-half row
      async_copy_b128(kb + (size_t)(kt + row) * QEXT + c8 * 8,
                      &ldsK[row * KSTR + c8 * 8]);
    }
#pragma unroll
    for (int i = 0; i < 2; ++i) {
      int c = tid + 256 * i;
      int row = c / 4, c8 = c % 4;       // 4 chunks of 8 halfs per 32-half row
      async_copy_b128(vb + (size_t)row * T_ + kt + c8 * 8,
                      &ldsV[row * VSTR + c8 * 8]);
    }
    wait_asynccnt0();
    __syncthreads();

    // ---- scores: 16 q-rows x 32 keys, K = 192 ----
    v8f s0 = zf, s1 = zf;
#pragma unroll
    for (int j = 0; j < 6; ++j) {
      v16h bk0 = load_frag_b(&ldsK[j * 32], KSTR);
      v16h bk1 = load_frag_b(&ldsK[16 * KSTR + j * 32], KSTR);
      s0 = __builtin_amdgcn_wmma_f32_16x16x32_f16(false, qf[j], false, bk0, (short)0, s0, false, false);
      s1 = __builtin_amdgcn_wmma_f32_16x16x32_f16(false, qf[j], false, bk1, (short)0, s1, false, false);
    }

    // ---- online softmax (rows live in half-wave lanes) ----
    FragF f0, f1; f0.v = s0; f1.v = s1;
#pragma unroll
    for (int r = 0; r < 8; ++r) {
      float mx = fmaxf(f0.f[r], f1.f[r]);
#pragma unroll
      for (int off = 8; off; off >>= 1) mx = fmaxf(mx, __shfl_xor(mx, off, 16));
      float mnew  = fmaxf(mrun[r], mx);
      float scale = __expf(mrun[r] - mnew);
      float p0 = __expf(f0.f[r] - mnew);
      float p1 = __expf(f1.f[r] - mnew);
      float ps = p0 + p1;
#pragma unroll
      for (int off = 8; off; off >>= 1) ps += __shfl_xor(ps, off, 16);
      lrun[r] = lrun[r] * scale + ps;
      mrun[r] = mnew;
#pragma unroll
      for (int n = 0; n < 8; ++n) acc[n][r] *= scale;
      int row = r + 8 * halfL;
      Pl[row * 32 + colL]      = (_Float16)p0;
      Pl[row * 32 + 16 + colL] = (_Float16)p1;
    }
    asm volatile("s_wait_dscnt 0" ::: "memory");  // wave-local LDS RAW fence
    v16h pf = load_frag_a(Pl, 32);                // re-layout P as A-fragment

    // ---- P(16x32) @ V(32x128) from LDS ----
#pragma unroll
    for (int n = 0; n < 8; ++n) {
      v16h bv = load_frag_b(&ldsV[(n * 16) * VSTR], VSTR);
      acc[n] = __builtin_amdgcn_wmma_f32_16x16x32_f16(false, pf, false, bv, (short)0, acc[n], false, false);
    }
    __syncthreads();   // all waves done with ldsK/ldsV before next stage
  }

  // normalize and write y[b, t, h*HD + d]
#pragma unroll
  for (int r = 0; r < 8; ++r) {
    int trow = q0 + r + 8 * halfL;
    float inv = 1.0f / lrun[r];
    float* yrow = y + ((size_t)b * T_ + trow) * D_ + h * HD;
#pragma unroll
    for (int n = 0; n < 8; ++n) {
      FragF a; a.v = acc[n];
      yrow[n * 16 + colL] = a.f[r] * inv;
    }
  }
}

extern "C" void kernel_launch(void* const* d_in, const int* in_sizes, int n_in,
                              void* d_out, int out_size, void* d_ws, size_t ws_size,
                              hipStream_t stream) {
  const float* x        = (const float*)d_in[0];
  const float* coords   = (const float*)d_in[1];
  const unsigned char* mask = (const unsigned char*)d_in[2];
  const float* Wqkv     = (const float*)d_in[3];
  const float* bqkv     = (const float*)d_in[4];
  const float* Wo       = (const float*)d_in[5];
  const float* bo       = (const float*)d_in[6];
  const float* W_rope   = (const float*)d_in[7];
  const float* W_fb     = (const float*)d_in[8];
  const float* beta_cos = (const float*)d_in[9];
  const float* beta_sin = (const float*)d_in[10];
  float* out = (float*)d_out;

  char* ws = (char*)d_ws;
  size_t off = 0;
  auto alloc = [&](size_t bytes) -> void* {
    void* p = ws + off;
    off += (bytes + 255) & ~(size_t)255;
    return p;
  };
  _Float16* WqkvT = (_Float16*)alloc((size_t)3 * D_ * D_ * 2);   // (3D x D) f16
  _Float16* WoT   = (_Float16*)alloc((size_t)D_ * D_ * 2);       // (D x D) f16
  float*    qkv   = (float*)alloc((size_t)BT * 3 * D_ * 4);      // (BT x 3D) f32
  _Float16* q_ext = (_Float16*)alloc((size_t)B_ * NH * T_ * QEXT * 2);
  _Float16* k_ext = (_Float16*)alloc((size_t)B_ * NH * T_ * QEXT * 2);
  _Float16* v_t   = (_Float16*)alloc((size_t)B_ * NH * HD * T_ * 2);
  float*    yattn = (float*)alloc((size_t)BT * D_ * 4);

  // 1) weights -> transposed f16
  {
    int tot = 3 * D_ * D_;
    transpose_f16_kernel<<<(tot + 255) / 256, 256, 0, stream>>>(Wqkv, WqkvT, D_, 3 * D_);
    tot = D_ * D_;
    transpose_f16_kernel<<<(tot + 255) / 256, 256, 0, stream>>>(Wo, WoT, D_, D_);
  }
  // 2) QKV projection: (4096 x 1024) * (1024 x 3072)
  gemm_wmma_kernel<0><<<dim3(3 * D_ / 64, BT / 64), 128, 0, stream>>>(
      x, WqkvT, bqkv, qkv, BT, 3 * D_, D_, nullptr, nullptr);
  // 3) RoPE rotate + fourier-feature append + V transpose
  rope_extend_kernel<<<BT, 64, 0, stream>>>(qkv, coords, W_rope, W_fb,
                                            beta_cos, beta_sin, q_ext, k_ext, v_t);
  // 4) flash attention with folded bias, async K/V staging
  flash_attn_kernel<<<dim3(B_ * NH, T_ / 128), 256, 0, stream>>>(q_ext, k_ext, v_t, yattn);
  // 5) output projection + masked residual select
  gemm_wmma_kernel<1><<<dim3(D_ / 64, BT / 64), 128, 0, stream>>>(
      yattn, WoT, bo, out, BT, D_, D_, mask, x);
}
